// ContextTokenModel_73864847557027
// MI455X (gfx1250) — compile-verified, matrix operand order checked by reading, not run on hardware
//
#include <hip/hip_runtime.h>
#include <hip/hip_bf16.h>

// ---------------------------------------------------------------------------
// ContextTokenModel on MI455X (gfx1250): embedding gather + 2-layer bidir GRU
// All GEMMs via v_wmma_f32_16x16x32_bf16 (bf16 in, f32 accumulate).
// ---------------------------------------------------------------------------

typedef __attribute__((ext_vector_type(16))) __bf16 v16bf;
typedef __attribute__((ext_vector_type(8)))  __bf16 bf16x8;
typedef __attribute__((ext_vector_type(8)))  float  v8f;

// Load one 16x32 bf16 WMMA operand fragment for this lane.
// Layout (CDNA5 ISA 7.12.2, 16-bit A 16x32): lanes 0-15 hold K[0..7],K[16..23]
// of row M=lane; lanes 16-31 hold K[8..15],K[24..31] of row M=lane-16.
// `p` points at (row_base + kbase); koff = (lane>>4)*8.
static __device__ __forceinline__ v16bf ld_frag(const __bf16* p, int koff) {
  bf16x8 lo = *reinterpret_cast<const bf16x8*>(p + koff);
  bf16x8 hi = *reinterpret_cast<const bf16x8*>(p + 16 + koff);
  v16bf f;
#pragma unroll
  for (int i = 0; i < 8; ++i) { f[i] = lo[i]; f[8 + i] = hi[i]; }
  return f;
}

static __device__ __forceinline__ v8f wmma_bf16(v16bf a, v16bf b, v8f c) {
  // (neg_a, A, neg_b, B, c_mod, C, reuse_a, reuse_b)
  return __builtin_amdgcn_wmma_f32_16x16x32_bf16(false, a, false, b, (short)0, c,
                                                 false, false);
}

// ---------------------------------------------------------------------------
// Stage 1: build X[t][b][160] (bf16) = concat(token_emb, masked-max type_emb),
// with t==10 being the broadcast hole representation.
// ---------------------------------------------------------------------------
__global__ __launch_bounds__(160) void embed_kernel(
    const int* __restrict__ tok_b, const int* __restrict__ typ_b,
    const unsigned char* __restrict__ msk_b,
    const int* __restrict__ tok_a, const int* __restrict__ typ_a,
    const unsigned char* __restrict__ msk_a,
    const float* __restrict__ temb, const float* __restrict__ yemb,
    const float* __restrict__ hole, __bf16* __restrict__ X, int Bn) {
  const int b = blockIdx.x;
  const int t = blockIdx.y;     // 0..20
  const int d = threadIdx.x;    // 0..159
  float val;
  if (t == 10) {
    val = hole[d];
  } else {
    const int st = (t < 10) ? t : (t - 11);
    const int* tok = (t < 10) ? tok_b : tok_a;
    const int* typ = (t < 10) ? typ_b : typ_a;
    const unsigned char* msk = (t < 10) ? msk_b : msk_a;
    if (d < 128) {
      val = temb[(size_t)tok[b * 10 + st] * 128 + d];
    } else {
      const int j = d - 128;
      float mx = -3.4e38f;
#pragma unroll
      for (int i = 0; i < 10; ++i) {
        const int base = (b * 10 + st) * 10 + i;
        float v = yemb[(size_t)typ[base] * 32 + j];
        if (!msk[base]) v -= 1000.0f;   // masked-out slot: shift way down
        mx = fmaxf(mx, v);
      }
      val = mx;
    }
  }
  X[((size_t)t * Bn + b) * 160 + d] = (__bf16)val;
}

// ---------------------------------------------------------------------------
// Stage 2/3: one bidirectional GRU layer. Each wave owns 16 batch rows and
// scans all 21 steps privately (no grid sync needed; recurrence is batch-
// parallel). Weights live in LDS as bf16, N-major for contiguous K loads.
// Per step:  ru = sigmoid([x,h]@gk + gb); c = tanh([x, r*h]@ck + cb);
//            h' = u*h + (1-u)*c.
// ---------------------------------------------------------------------------
template <int DIN>
__global__ __launch_bounds__(128) void gru_layer_kernel(
    const __bf16* __restrict__ X, int Bn,
    const float* __restrict__ gk_fw, const float* __restrict__ gb_fw,
    const float* __restrict__ ck_fw, const float* __restrict__ cb_fw,
    const float* __restrict__ gk_bw, const float* __restrict__ gb_bw,
    const float* __restrict__ ck_bw, const float* __restrict__ cb_bw,
    __bf16* __restrict__ Ybf,     // layer0: [t][b][64] bf16 (fw cols 0-31)
    float* __restrict__ Yf32) {   // layer1: [b][21][64] f32 final output
  constexpr int K   = DIN + 32;   // [x,h] concat depth
  constexpr int KP  = K + 8;      // pad: rows 16B-aligned, bank-skewed
  constexpr int XKT = DIN / 32;   // K-tiles covering the x part

  __shared__ __bf16 wg[64 * KP];        // gate kernel, N-major [n][k]
  __shared__ __bf16 wc[32 * KP];        // candidate kernel, N-major
  __shared__ __bf16 hs[4 * 16 * 32];    // per-wave h (row-major)
  __shared__ __bf16 rhs_[4 * 16 * 32];  // per-wave r*h staging

  const int tid = threadIdx.x;
  const int dir = blockIdx.y;           // 0 = fw, 1 = bw
  const float* gk = dir ? gk_bw : gk_fw;
  const float* gb = dir ? gb_bw : gb_fw;
  const float* ck = dir ? ck_bw : ck_fw;
  const float* cb = dir ? cb_bw : cb_fw;

  // Stage weights fp32 -> bf16, transposed to [n][k].
  for (int i = tid; i < 64 * K; i += 128) {
    const int n = i / K, k = i - n * K;
    wg[n * KP + k] = (__bf16)gk[k * 64 + n];
  }
  for (int i = tid; i < 32 * K; i += 128) {
    const int n = i / K, k = i - n * K;
    wc[n * KP + k] = (__bf16)ck[k * 32 + n];
  }
  __syncthreads();

  const int wave  = tid >> 5;
  const int lane  = tid & 31;
  const int m     = lane & 15;            // A-row / accumulator N index
  const int koff  = (lane >> 4) << 3;     // fragment K sub-offset (0 or 8)
  const int mrow2 = (lane >> 4) << 3;     // accumulator M offset (0 or 8)
  const int b0    = blockIdx.x * 64 + wave * 16;

  __bf16* hw  = &hs[wave * 512];
  __bf16* rhw = &rhs_[wave * 512];
  if (lane < 16) {
#pragma unroll
    for (int c = 0; c < 32; ++c) hw[m * 32 + c] = (__bf16)0.0f;
  }

  float gB[4], cB[2];
#pragma unroll
  for (int nt = 0; nt < 4; ++nt) gB[nt] = gb[nt * 16 + m];
#pragma unroll
  for (int nt = 0; nt < 2; ++nt) cB[nt] = cb[nt * 16 + m];

  for (int s = 0; s < 21; ++s) {
    const int t = dir ? (20 - s) : s;
    const __bf16* xrow = X + ((size_t)t * Bn + (b0 + m)) * DIN;

    v8f aG[4], aC[2];
#pragma unroll
    for (int nt = 0; nt < 4; ++nt)
#pragma unroll
      for (int v = 0; v < 8; ++v) aG[nt][v] = gB[nt];
#pragma unroll
    for (int nt = 0; nt < 2; ++nt)
#pragma unroll
      for (int v = 0; v < 8; ++v) aC[nt][v] = cB[nt];

    // x contribution for gates AND candidate (fused K-loop; X slice is L2-hot)
#pragma unroll
    for (int kt = 0; kt < XKT; ++kt) {
      const v16bf a = ld_frag(xrow + kt * 32, koff);
#pragma unroll
      for (int nt = 0; nt < 4; ++nt)
        aG[nt] = wmma_bf16(a, ld_frag(&wg[(nt * 16 + m) * KP + kt * 32], koff), aG[nt]);
#pragma unroll
      for (int nt = 0; nt < 2; ++nt)
        aC[nt] = wmma_bf16(a, ld_frag(&wc[(nt * 16 + m) * KP + kt * 32], koff), aC[nt]);
    }

    // recurrent gate contribution: h @ gk[DIN:DIN+32]
    {
      const v16bf ah = ld_frag(&hw[m * 32], koff);
#pragma unroll
      for (int nt = 0; nt < 4; ++nt)
        aG[nt] = wmma_bf16(ah, ld_frag(&wg[(nt * 16 + m) * KP + DIN], koff), aG[nt]);
    }

    // ru = sigmoid(...)  (r = cols 0..31, u = cols 32..63)
#pragma unroll
    for (int nt = 0; nt < 4; ++nt)
#pragma unroll
      for (int v = 0; v < 8; ++v)
        aG[nt][v] = 1.0f / (1.0f + __expf(-aG[nt][v]));

    // stage r*h back through LDS into A-matrix layout
#pragma unroll
    for (int nt = 0; nt < 2; ++nt)
#pragma unroll
      for (int v = 0; v < 8; ++v) {
        const int mm = v + mrow2;
        const int cc = nt * 16 + m;
        const float hv = (float)hw[mm * 32 + cc];
        rhw[mm * 32 + cc] = (__bf16)(aG[nt][v] * hv);
      }

    // candidate recurrent contribution: (r*h) @ ck[DIN:DIN+32]
    {
      const v16bf arh = ld_frag(&rhw[m * 32], koff);
#pragma unroll
      for (int nt = 0; nt < 2; ++nt)
        aC[nt] = wmma_bf16(arh, ld_frag(&wc[(nt * 16 + m) * KP + DIN], koff), aC[nt]);
    }

    // h' = u*h + (1-u)*tanh(c); write new h + layer output
#pragma unroll
    for (int nt = 0; nt < 2; ++nt)
#pragma unroll
      for (int v = 0; v < 8; ++v) {
        const int mm = v + mrow2;
        const int cc = nt * 16 + m;
        const float c  = tanhf(aC[nt][v]);
        const float u  = aG[2 + nt][v];
        const float hv = (float)hw[mm * 32 + cc];
        const float hn = u * hv + (1.0f - u) * c;
        hw[mm * 32 + cc] = (__bf16)hn;
        if (Ybf) {
          Ybf[((size_t)t * Bn + (b0 + mm)) * 64 + dir * 32 + cc] = (__bf16)hn;
        } else {
          Yf32[((size_t)(b0 + mm) * 21 + t) * 64 + dir * 32 + cc] = hn;
        }
      }
  }
}

// ---------------------------------------------------------------------------
extern "C" void kernel_launch(void* const* d_in, const int* in_sizes, int n_in,
                              void* d_out, int out_size, void* d_ws, size_t ws_size,
                              hipStream_t stream) {
  const int* tok_b = (const int*)d_in[0];
  const int* typ_b = (const int*)d_in[1];
  const unsigned char* msk_b = (const unsigned char*)d_in[2];
  const int* tok_a = (const int*)d_in[3];
  const int* typ_a = (const int*)d_in[4];
  const unsigned char* msk_a = (const unsigned char*)d_in[5];
  const float* temb = (const float*)d_in[6];
  const float* yemb = (const float*)d_in[7];
  const float* hole = (const float*)d_in[8];
  const float* w[16];
  for (int i = 0; i < 16; ++i) w[i] = (const float*)d_in[9 + i];

  const int B = in_sizes[0] / 10;  // 8192 (multiple of 64)

  // Workspace: X [21][B][160] bf16, then y0 [21][B][64] bf16.
  __bf16* X = (__bf16*)d_ws;
  size_t xbytes = ((size_t)21 * B * 160 * sizeof(__bf16) + 255) & ~(size_t)255;
  __bf16* Y0 = (__bf16*)((char*)d_ws + xbytes);

  embed_kernel<<<dim3(B, 21), 160, 0, stream>>>(
      tok_b, typ_b, msk_b, tok_a, typ_a, msk_a, temb, yemb, hole, X, B);

  gru_layer_kernel<160><<<dim3(B / 64, 2), 128, 0, stream>>>(
      X, B, w[0], w[1], w[2], w[3], w[4], w[5], w[6], w[7], Y0, nullptr);

  gru_layer_kernel<64><<<dim3(B / 64, 2), 128, 0, stream>>>(
      Y0, B, w[8], w[9], w[10], w[11], w[12], w[13], w[14], w[15],
      nullptr, (float*)d_out);
}
